// DGCNN_Cls_Encoder_80874234183926
// MI455X (gfx1250) — matrix operand order repaired and checked by reading, not compiled, not run on hardware
//
#include <hip/hip_runtime.h>
#include <hip/hip_bf16.h>
#include <math.h>

typedef __attribute__((ext_vector_type(16))) _Float16 v16h;
typedef __attribute__((ext_vector_type(8)))  float    v8f;

#define BB 8
#define NN 2048
#define KNB 20

// ---------------- weight convert (f32 -> f16, zero-padded cols) ----------------
__global__ void k_cvt_w(const float* __restrict__ src, _Float16* __restrict__ dst,
                        int rows, int cs, int cd) {
  int i = blockIdx.x * blockDim.x + threadIdx.x;
  if (i >= rows * cd) return;
  int r = i / cd, c = i % cd;
  dst[i] = (_Float16)(c < cs ? src[r * cs + c] : 0.0f);
}

// ---------------- zero stats + pooled init ----------------
__global__ void k_init(float* stats2048, unsigned* pooled_u) {
  int i = blockIdx.x * blockDim.x + threadIdx.x;
  if (i < 2048) stats2048[i] = 0.0f;
  if (i < 4096) pooled_u[i] = 0u;   // ordered-uint 0 == most negative
}

// ---------------- per-point squared norm ----------------
__global__ void k_xx(const float* __restrict__ X, int ldX, int coff, int C,
                     float* __restrict__ xx, int total) {
  int i = blockIdx.x * blockDim.x + threadIdx.x;
  if (i >= total) return;
  const float* p = X + (size_t)i * ldX + coff;
  float s = 0.0f;
  for (int c = 0; c < C; ++c) s += p[c] * p[c];
  xx[i] = s;
}

// ---------------- kNN: top-20 of d = 2*dot - xxq - xxc (stable ties) ----------------
__global__ __launch_bounds__(64) void k_knn(const float* __restrict__ X, int ldX, int coff,
                                            int C, const float* __restrict__ xx,
                                            int* __restrict__ idxout, int N) {
  __shared__ float qb[64][129];
  __shared__ float cb[32][128];
  int b = blockIdx.y, n0 = blockIdx.x * 64, t = threadIdx.x;
  int n = n0 + t;
  size_t bN = (size_t)b * N;
  const float* qp = X + (bN + n) * ldX + coff;
  for (int c = 0; c < C; ++c) qb[t][c] = qp[c];
  float xxq = xx[bN + n];
  float dl[KNB]; int il[KNB];
  for (int p = 0; p < KNB; ++p) { dl[p] = -3.4e38f; il[p] = 0x7fffffff; }
  for (int c0 = 0; c0 < N; c0 += 32) {
    __syncthreads();
    for (int e = t; e < 32 * C; e += 64) {
      int j = e / C, c = e % C;
      cb[j][c] = X[(bN + c0 + j) * ldX + coff + c];
    }
    __syncthreads();
    for (int j = 0; j < 32; ++j) {
      float dot = 0.0f;
      for (int c = 0; c < C; ++c) dot += qb[t][c] * cb[j][c];
      float d = 2.0f * dot - xxq - xx[bN + c0 + j];
      if (d > dl[KNB - 1]) {       // strict > => stable order like top_k
        int p = KNB - 1;
        while (p > 0 && d > dl[p - 1]) { dl[p] = dl[p - 1]; il[p] = il[p - 1]; --p; }
        dl[p] = d; il[p] = c0 + j;
      }
    }
  }
  int* op = idxout + (bN + n) * KNB;
  for (int p = 0; p < KNB; ++p) op[p] = il[p];
}

// ---------------- edge conv GEMM via WMMA (PASS 0: stats, PASS 1: apply) ----------------
// cols = 8 queries * 20 neighbors = 160 (10 col-tiles of 16); rows = 64 per block.
template <int PASS>
__global__ __launch_bounds__(128) void k_edge(
    const float* __restrict__ Xin, int ldX, int coff, int C, int twoC, int Kpad,
    const _Float16* __restrict__ W, const int* __restrict__ idx, int N,
    float* __restrict__ sums, float* __restrict__ sumsqs,
    const float* __restrict__ sc, const float* __restrict__ sh,
    float* __restrict__ xcat, int coffO) {
  __shared__ _Float16 Fs[160][32];
  __shared__ float    Ys[64][164];
  __shared__ int      s_nbr[160];
  int b = blockIdx.z, n0 = blockIdx.x * 8, o0 = blockIdx.y * 64;
  int t = threadIdx.x, lane = t & 31, wid = t >> 5, h = lane >> 4;
  size_t bN = (size_t)b * N;
  if (t < 160) s_nbr[t] = idx[(bN + n0 + t / KNB) * KNB + (t % KNB)];
  v8f acc[10];
  v8f zero = {};
#pragma unroll
  for (int ct = 0; ct < 10; ++ct) acc[ct] = zero;
  int m = o0 + wid * 16 + (lane & 15);
  const _Float16* wrow = W + (size_t)m * Kpad;

  for (int kk0 = 0; kk0 < Kpad; kk0 += 32) {
    __syncthreads();
    for (int e = t; e < 160 * 32; e += 128) {
      int col = e >> 5, kk = e & 31, c2 = kk0 + kk;
      float v = 0.0f;
      if (c2 < twoC) {
        int n = n0 + col / KNB;
        if (c2 < C)
          v = Xin[(bN + s_nbr[col]) * ldX + coff + c2] - Xin[(bN + n) * ldX + coff + c2];
        else
          v = Xin[(bN + n) * ldX + coff + (c2 - C)];
      }
      Fs[col][kk] = (_Float16)v;
    }
    __syncthreads();
    if (kk0 + 32 < Kpad) __builtin_prefetch(wrow + kk0 + 32, 0, 3);
    v16h a;
    {
      const _Float16* p0 = wrow + kk0 + h * 16;
#pragma unroll
      for (int i = 0; i < 8; ++i) { a[i] = p0[i]; a[i + 8] = p0[8 + i]; }
    }
    // preload all B fragments, then issue WMMAs back-to-back
    v16h bf[10];
#pragma unroll
    for (int ct = 0; ct < 10; ++ct)
      bf[ct] = *reinterpret_cast<const v16h*>(&Fs[ct * 16 + (lane & 15)][h * 16]);
#pragma unroll
    for (int ct = 0; ct < 10; ++ct)
      acc[ct] = __builtin_amdgcn_wmma_f32_16x16x32_f16(false, a, false, bf[ct],
                                                       (short)0, acc[ct], false, false);
  }
#pragma unroll
  for (int ct = 0; ct < 10; ++ct)
#pragma unroll
    for (int r = 0; r < 8; ++r)
      Ys[wid * 16 + 8 * h + r][ct * 16 + (lane & 15)] = acc[ct][r];
  __syncthreads();

  if (t < 64) {
    int o = o0 + t;
    if (PASS == 0) {
      float s = 0.0f, s2 = 0.0f;
      for (int col = 0; col < 160; ++col) { float v = Ys[t][col]; s += v; s2 += v * v; }
      atomicAdd(&sums[o], s);
      atomicAdd(&sumsqs[o], s2);
    } else {
      float a_ = sc[o], b_ = sh[o];
      for (int q = 0; q < 8; ++q) {
        float mx = -3.4e38f;
        for (int k = 0; k < KNB; ++k) {
          float v = Ys[t][q * KNB + k] * a_ + b_;
          v = v > 0.0f ? v : 0.2f * v;
          mx = fmaxf(mx, v);
        }
        xcat[(bN + n0 + q) * 512 + coffO + o] = mx;
      }
    }
  }
}

// ---------------- BN finalize: mean/var -> scale/shift ----------------
__global__ void k_bnfin(const float* __restrict__ sums, const float* __restrict__ sumsqs,
                        const float* __restrict__ g, const float* __restrict__ bb,
                        float* __restrict__ sc, float* __restrict__ sh, float cnt, int O) {
  int o = blockIdx.x * blockDim.x + threadIdx.x;
  if (o >= O) return;
  float mean = sums[o] / cnt;
  float var = sumsqs[o] / cnt - mean * mean;
  float inv = rsqrtf(var + 1e-5f);
  float s = g[o] * inv;
  sc[o] = s;
  sh[o] = bb[o] - mean * s;
}

// ---------------- final 512x512 conv via WMMA ----------------
template <int PASS>
__global__ __launch_bounds__(128) void k_final(
    const float* __restrict__ xcat, const _Float16* __restrict__ W5, int N,
    float* __restrict__ sums, float* __restrict__ sumsqs,
    const float* __restrict__ sc, const float* __restrict__ sh,
    unsigned* __restrict__ pooled_u) {
  __shared__ _Float16 Fs[128][32];
  __shared__ float    Ys[64][132];
  int b = blockIdx.z, n0 = blockIdx.x * 128, o0 = blockIdx.y * 64;
  int t = threadIdx.x, lane = t & 31, wid = t >> 5, h = lane >> 4;
  size_t bN = (size_t)b * N;
  v8f acc[8];
  v8f zero = {};
#pragma unroll
  for (int ct = 0; ct < 8; ++ct) acc[ct] = zero;
  int m = o0 + wid * 16 + (lane & 15);
  const _Float16* wrow = W5 + (size_t)m * 512;

  for (int kk0 = 0; kk0 < 512; kk0 += 32) {
    __syncthreads();
    for (int e = t; e < 128 * 32; e += 128) {
      int col = e >> 5, kk = e & 31;
      Fs[col][kk] = (_Float16)xcat[(bN + n0 + col) * 512 + kk0 + kk];
    }
    __syncthreads();
    if (kk0 + 32 < 512) __builtin_prefetch(wrow + kk0 + 32, 0, 3);
    v16h a;
    {
      const _Float16* p0 = wrow + kk0 + h * 16;
#pragma unroll
      for (int i = 0; i < 8; ++i) { a[i] = p0[i]; a[i + 8] = p0[8 + i]; }
    }
    v16h bf[8];
#pragma unroll
    for (int ct = 0; ct < 8; ++ct)
      bf[ct] = *reinterpret_cast<const v16h*>(&Fs[ct * 16 + (lane & 15)][h * 16]);
#pragma unroll
    for (int ct = 0; ct < 8; ++ct)
      acc[ct] = __builtin_amdgcn_wmma_f32_16x16x32_f16(false, a, false, bf[ct],
                                                       (short)0, acc[ct], false, false);
  }
#pragma unroll
  for (int ct = 0; ct < 8; ++ct)
#pragma unroll
    for (int r = 0; r < 8; ++r)
      Ys[wid * 16 + 8 * h + r][ct * 16 + (lane & 15)] = acc[ct][r];
  __syncthreads();

  if (t < 64) {
    int o = o0 + t;
    if (PASS == 0) {
      float s = 0.0f, s2 = 0.0f;
      for (int col = 0; col < 128; ++col) { float v = Ys[t][col]; s += v; s2 += v * v; }
      atomicAdd(&sums[o], s);
      atomicAdd(&sumsqs[o], s2);
    } else {
      float a_ = sc[o], b_ = sh[o];
      float mx = -3.4e38f;
      for (int col = 0; col < 128; ++col) {
        float v = Ys[t][col] * a_ + b_;
        v = v > 0.0f ? v : 0.2f * v;
        mx = fmaxf(mx, v);
      }
      unsigned u = __float_as_uint(mx);
      u = (u & 0x80000000u) ? ~u : (u | 0x80000000u);
      atomicMax(&pooled_u[(size_t)b * 512 + o], u);
    }
  }
}

// ---------------- decode pooled max + write feat output ----------------
__global__ void k_pool(const unsigned* __restrict__ pu, float* __restrict__ pooled,
                       float* __restrict__ outfeat) {
  int i = blockIdx.x * blockDim.x + threadIdx.x;
  if (i >= BB * 512) return;
  unsigned u = pu[i];
  float f = (u >> 31) ? __uint_as_float(u & 0x7fffffffu) : __uint_as_float(~u);
  pooled[i] = f;
  outfeat[i] = f;
}

// ---------------- embedding: pooled (B,512) @ w_emb^T (128,512) ----------------
__global__ void k_embed(const float* __restrict__ pooled, const float* __restrict__ wemb,
                        float* __restrict__ emb, float* __restrict__ outemb) {
  int i = blockIdx.x * blockDim.x + threadIdx.x;
  if (i >= BB * 128) return;
  int b = i / 128, e = i % 128;
  float s = 0.0f;
  const float* p = pooled + (size_t)b * 512;
  const float* w = wemb + (size_t)e * 512;
  for (int c = 0; c < 512; ++c) s += p[c] * w[c];
  emb[i] = s;
  outemb[i] = s;
}

// ---------------- clustering head ----------------
__global__ void k_cluster(const float* __restrict__ emb, const float* __restrict__ wclu,
                          float* __restrict__ outq) {
  __shared__ float qv[10];
  int b = blockIdx.x, t = threadIdx.x;
  if (t < 10) {
    float d2 = 0.0f;
    for (int c = 0; c < 128; ++c) {
      float d = emb[(size_t)b * 128 + c] - wclu[(size_t)t * 128 + c];
      d2 += d * d;
    }
    qv[t] = 1.0f / (1.0f + d2);
  }
  __syncthreads();
  if (t < 10) {
    float s = 0.0f;
    for (int j = 0; j < 10; ++j) s += qv[j];
    outq[(size_t)b * 10 + t] = qv[t] / s;
  }
}

extern "C" void kernel_launch(void* const* d_in, const int* in_sizes, int n_in,
                              void* d_out, int out_size, void* d_ws, size_t ws_size,
                              hipStream_t stream) {
  (void)in_sizes; (void)n_in; (void)out_size; (void)ws_size;
  const int B = BB, N = NN;
  const float* x = (const float*)d_in[0];
  float* dout = (float*)d_out;

  // ---- workspace bump allocation ----
  char* w = (char*)d_ws;
  auto take = [&](size_t bytes) { char* p = w; w += (bytes + 255) & ~(size_t)255; return p; };
  float*    xcat  = (float*)take((size_t)B * N * 512 * 4);
  int*      idxb  = (int*)take((size_t)B * N * KNB * 4);
  float*    xx    = (float*)take((size_t)B * N * 4);
  _Float16* wh    = (_Float16*)take((size_t)354304 * 2);
  float*    sums  = (float*)take(1024 * 4);
  float*    sumsq = (float*)take(1024 * 4);
  float*    sc    = (float*)take(1024 * 4);
  float*    sh    = (float*)take(1024 * 4);
  unsigned* pou   = (unsigned*)take((size_t)B * 512 * 4);
  float*    pooled= (float*)take((size_t)B * 512 * 4);
  float*    emb   = (float*)take((size_t)B * 128 * 4);

  _Float16* w1h = wh;
  _Float16* w2h = wh + 2048;
  _Float16* w3h = wh + 10240;
  _Float16* w4h = wh + 26624;
  _Float16* w5h = wh + 92160;

  // ---- weight conversion (f32 -> f16, layer-1 padded to K=32) ----
  k_cvt_w<<<dim3((64 * 32 + 255) / 256), 256, 0, stream>>>((const float*)d_in[1],  w1h, 64, 6, 32);
  k_cvt_w<<<dim3((64 * 128 + 255) / 256), 256, 0, stream>>>((const float*)d_in[4],  w2h, 64, 128, 128);
  k_cvt_w<<<dim3((128 * 128 + 255) / 256), 256, 0, stream>>>((const float*)d_in[7],  w3h, 128, 128, 128);
  k_cvt_w<<<dim3((256 * 256 + 255) / 256), 256, 0, stream>>>((const float*)d_in[10], w4h, 256, 256, 256);
  k_cvt_w<<<dim3((512 * 512 + 255) / 256), 256, 0, stream>>>((const float*)d_in[13], w5h, 512, 512, 512);

  k_init<<<dim3(16), 256, 0, stream>>>(sums, pou);  // sums+sumsq contiguous 2048 floats

  // ---- edge block descriptors ----
  struct L { const float* Xin; int ldX, coff, C, twoC, Kpad, O, so, coffO; const float* g; const float* b; const _Float16* W; };
  L Ls[4] = {
      {x,    3,   0,   3,   6,   32,  64,  0,   0,   (const float*)d_in[2],  (const float*)d_in[3],  w1h},
      {xcat, 512, 0,   64,  128, 128, 64,  64,  64,  (const float*)d_in[5],  (const float*)d_in[6],  w2h},
      {xcat, 512, 64,  64,  128, 128, 128, 128, 128, (const float*)d_in[8],  (const float*)d_in[9],  w3h},
      {xcat, 512, 128, 128, 256, 256, 256, 256, 256, (const float*)d_in[11], (const float*)d_in[12], w4h},
  };
  const float cnt_edge = (float)B * N * KNB;

  for (int l = 0; l < 4; ++l) {
    const L& a = Ls[l];
    k_xx<<<dim3((B * N + 255) / 256), 256, 0, stream>>>(a.Xin, a.ldX, a.coff, a.C, xx, B * N);
    k_knn<<<dim3(N / 64, B), 64, 0, stream>>>(a.Xin, a.ldX, a.coff, a.C, xx, idxb, N);
    dim3 g(N / 8, a.O / 64, B);
    k_edge<0><<<g, 128, 0, stream>>>(a.Xin, a.ldX, a.coff, a.C, a.twoC, a.Kpad, a.W, idxb, N,
                                     sums + a.so, sumsq + a.so, sc + a.so, sh + a.so, xcat, a.coffO);
    k_bnfin<<<dim3((a.O + 63) / 64), 64, 0, stream>>>(sums + a.so, sumsq + a.so, a.g, a.b,
                                                      sc + a.so, sh + a.so, cnt_edge, a.O);
    k_edge<1><<<g, 128, 0, stream>>>(a.Xin, a.ldX, a.coff, a.C, a.twoC, a.Kpad, a.W, idxb, N,
                                     sums + a.so, sumsq + a.so, sc + a.so, sh + a.so, xcat, a.coffO);
  }

  // ---- final conv + BN + relu + global max pool ----
  dim3 gf(N / 128, 512 / 64, B);
  k_final<0><<<gf, 128, 0, stream>>>(xcat, w5h, N, sums + 512, sumsq + 512, sc + 512, sh + 512, pou);
  k_bnfin<<<dim3(8), 64, 0, stream>>>(sums + 512, sumsq + 512, (const float*)d_in[14],
                                      (const float*)d_in[15], sc + 512, sh + 512, (float)(B * N), 512);
  k_final<1><<<gf, 128, 0, stream>>>(xcat, w5h, N, sums + 512, sumsq + 512, sc + 512, sh + 512, pou);

  // ---- heads ----
  k_pool<<<dim3(16), 256, 0, stream>>>(pou, pooled, dout);                       // feat (B,1,512)
  k_embed<<<dim3(4), 256, 0, stream>>>(pooled, (const float*)d_in[16], emb, dout + 4096);
  k_cluster<<<dim3(B), 32, 0, stream>>>(emb, (const float*)d_in[17], dout + 4096 + 1024);
}
